// FlowNet_77558519432067
// MI455X (gfx1250) — matrix operand order repaired and checked by reading, not compiled, not run on hardware
//
#include <hip/hip_runtime.h>

typedef __attribute__((ext_vector_type(16))) _Float16 v16h;
typedef __attribute__((ext_vector_type(8)))  float    v8f;

#define Bn   16
#define CIN  7
#define CH   32
#define S    256          // scan length
#define NI   256          // inner (conv) length
#define HWSZ (256*256)
#define CHW  (CH*HWSZ)
#define KTOT 160          // 32 c_in * 5 taps
#define LDW  (NI+4)       // carry row width incl. 2+2 zero pad

// ---------------------------------------------------------------------------
// Directional scan: one block per batch image, 512 threads = 16 wave32 waves.
// Each wave owns one M-tile (16 c_out) and TWO adjacent N-tiles (2 x 16 inner
// positions), so a single 40-VGPR A-set (weights as f16 WMMA A-tiles) feeds
// two accumulators -- fits the 4-waves/SIMD VGPR budget with no scratch
// spills. Carry row lives in zero-padded LDS; each step:
//   D = W(32x160) x im2col(carry)(160x256) via v_wmma_f32_16x16x32_f16,
//   new = x + tanh(D + bias) -> LDS carry + in-place Y.
// ---------------------------------------------------------------------------
__global__ __launch_bounds__(512)
void scan_kernel(float* __restrict__ Y, const float* __restrict__ wgt,
                 const float* __restrict__ bias, int st_t, int st_i, int rev)
{
    __shared__ float shC[CH * LDW];

    const int tid   = threadIdx.x;
    const int lane  = tid & 31;
    const int wave  = tid >> 5;          // 0..15
    const int half  = lane >> 4;         // 0 or 1
    const int n     = lane & 15;
    const int mtile = wave >> 3;         // 0..1  (c_out tile)
    const int npair = wave & 7;          // 0..7  -> ntiles 2*npair, 2*npair+1
    const int b     = blockIdx.x;
    const size_t base = (size_t)b * CHW;

    // ---- A-matrix (weights) for this wave's mtile: 5 K-tiles pinned in VGPRs.
    // 16-bit A 16x32 layout: elem e<8 -> K = half*8+e ; e>=8 -> K = 16+half*8+(e-8)
    v16h A[5];
    const int co_a = mtile * 16 + n;     // A row (M) = lane&15
    #pragma unroll
    for (int s = 0; s < 5; ++s) {
        #pragma unroll
        for (int e = 0; e < 16; ++e) {
            int K = s * 32 + ((e < 8) ? (half * 8 + e) : (16 + half * 8 + (e - 8)));
            A[s][e] = (_Float16)wgt[co_a * KTOT + K];   // K == c_in*5 + tap (flat)
        }
    }
    // bias per D row this lane owns: M = half*8 + v
    float bv[8];
    #pragma unroll
    for (int v = 0; v < 8; ++v)
        bv[v] = bias[mtile * 16 + half * 8 + v];

    // ---- initial carry = first scan line (zero-padded left/right by 2)
    {
        const int t0 = rev ? (S - 1) : 0;
        for (int idx = tid; idx < CH * LDW; idx += 512) {
            int c   = idx / LDW;
            int col = idx - c * LDW;
            float v = 0.f;
            if (col >= 2 && col < NI + 2)
                v = Y[base + (size_t)c * HWSZ + (size_t)t0 * st_t + (size_t)(col - 2) * st_i];
            shC[idx] = v;
        }
    }
    __syncthreads();

    const int col0 = (2 * npair) * 16 + n;       // first tile's inner position
    const int col1 = col0 + 16;                  // second tile's inner position

    for (int step = 1; step < S; ++step) {
        const int t = rev ? (S - 1 - step) : step;

        // ---- two 16x16 GEMM tiles: acc += A(16x32 f16) * B(32x16 f16), 5 K-tiles
        v8f acc0 = {}, acc1 = {};
        #pragma unroll
        for (int s = 0; s < 5; ++s) {
            // 16-bit B 32x16 layout: elem e<8 -> K = half*8+e ; e>=8 -> K = 16+half*8+(e-8), N = lane&15
            v16h Bm0, Bm1;
            #pragma unroll
            for (int e = 0; e < 16; ++e) {
                int K   = s * 32 + ((e < 8) ? (half * 8 + e) : (16 + half * 8 + (e - 8)));
                int ci  = K / 5;
                int tap = K - ci * 5;
                int off = ci * LDW + tap;        // constant per (s,e,half)
                // im2col with zero pad: memory col = col + tap - 2 -> LDS col + 2
                Bm0[e] = (_Float16)shC[off + col0];
                Bm1[e] = (_Float16)shC[off + col1];
            }
            acc0 = __builtin_amdgcn_wmma_f32_16x16x32_f16(
                       false, A[s], false, Bm0, (short)0, acc0, false, false);
            acc1 = __builtin_amdgcn_wmma_f32_16x16x32_f16(
                       false, A[s], false, Bm1, (short)0, acc1, false, false);
        }

        // ---- current line x (each lane reads exactly the elements it will write)
        float xv0[8], xv1[8];
        const size_t lidx0 = base + (size_t)t * st_t + (size_t)col0 * st_i;
        const size_t lidx1 = base + (size_t)t * st_t + (size_t)col1 * st_i;
        #pragma unroll
        for (int v = 0; v < 8; ++v) {
            int c = mtile * 16 + half * 8 + v;
            xv0[v] = Y[lidx0 + (size_t)c * HWSZ];
            xv1[v] = Y[lidx1 + (size_t)c * HWSZ];
        }
        __syncthreads();   // all reads of shC done before overwrite

        #pragma unroll
        for (int v = 0; v < 8; ++v) {
            int c = mtile * 16 + half * 8 + v;
            float nv0 = xv0[v] + tanhf(acc0[v] + bv[v]);
            float nv1 = xv1[v] + tanhf(acc1[v] + bv[v]);
            shC[c * LDW + col0 + 2] = nv0;       // next step's carry
            shC[c * LDW + col1 + 2] = nv1;
            Y[lidx0 + (size_t)c * HWSZ] = nv0;   // in-place output
            Y[lidx1 + (size_t)c * HWSZ] = nv1;
        }
        __syncthreads();   // new carry visible to all waves
    }
}

// ---------------------------------------------------------------------------
// Init conv: 3x3, 7 -> 32 channels, pad 1, tanh.  Weights staged in LDS.
// ---------------------------------------------------------------------------
__global__ __launch_bounds__(256)
void init_conv_kernel(const float* __restrict__ X, const float* __restrict__ W,
                      const float* __restrict__ bias, float* __restrict__ Y)
{
    __shared__ float sw[CH * CIN * 9];
    __shared__ float sb[CH];
    const int tid = threadIdx.x;
    for (int i = tid; i < CH * CIN * 9; i += 256) sw[i] = W[i];
    if (tid < CH) sb[tid] = bias[tid];
    __syncthreads();

    const size_t gid = (size_t)blockIdx.x * 256 + tid;   // over B*H*W
    const int w = (int)(gid & 255);
    const int h = (int)((gid >> 8) & 255);
    const int b = (int)(gid >> 16);

    float xr[CIN][9];
    #pragma unroll
    for (int ci = 0; ci < CIN; ++ci) {
        #pragma unroll
        for (int k = 0; k < 9; ++k) {
            int hh = h + k / 3 - 1, ww = w + k % 3 - 1;
            float v = 0.f;
            if (hh >= 0 && hh < 256 && ww >= 0 && ww < 256)
                v = X[((size_t)b * CIN + ci) * HWSZ + hh * 256 + ww];
            xr[ci][k] = v;
        }
    }
    for (int co = 0; co < CH; ++co) {
        float a = sb[co];
        const float* wp = &sw[co * CIN * 9];
        #pragma unroll
        for (int ci = 0; ci < CIN; ++ci)
            #pragma unroll
            for (int k = 0; k < 9; ++k)
                a = fmaf(xr[ci][k], wp[ci * 9 + k], a);
        Y[((size_t)b * CH + co) * HWSZ + h * 256 + w] = tanhf(a);
    }
}

// ---------------------------------------------------------------------------
// End conv: 3x3, 32 -> 1 channel, pad 1, sigmoid.
// ---------------------------------------------------------------------------
__global__ __launch_bounds__(256)
void end_conv_kernel(const float* __restrict__ Y, const float* __restrict__ W,
                     const float* __restrict__ bias, float* __restrict__ Out)
{
    __shared__ float sw[CH * 9];
    const int tid = threadIdx.x;
    for (int i = tid; i < CH * 9; i += 256) sw[i] = W[i];
    __syncthreads();

    const size_t gid = (size_t)blockIdx.x * 256 + tid;
    const int w = (int)(gid & 255);
    const int h = (int)((gid >> 8) & 255);
    const int b = (int)(gid >> 16);

    float a = bias[0];
    for (int ci = 0; ci < CH; ++ci) {
        const float* yp = &Y[((size_t)b * CH + ci) * HWSZ];
        #pragma unroll
        for (int dy = 0; dy < 3; ++dy) {
            int hh = h + dy - 1;
            if (hh < 0 || hh >= 256) continue;
            #pragma unroll
            for (int dx = 0; dx < 3; ++dx) {
                int ww = w + dx - 1;
                if (ww < 0 || ww >= 256) continue;
                a = fmaf(yp[hh * 256 + ww], sw[ci * 9 + dy * 3 + dx], a);
            }
        }
    }
    Out[gid] = 1.f / (1.f + __expf(-a));
}

// ---------------------------------------------------------------------------
extern "C" void kernel_launch(void* const* d_in, const int* in_sizes, int n_in,
                              void* d_out, int out_size, void* d_ws, size_t ws_size,
                              hipStream_t stream)
{
    (void)in_sizes; (void)n_in; (void)out_size; (void)ws_size;

    const float* X      = (const float*)d_in[0];
    const float* w_init = (const float*)d_in[1];
    const float* b_init = (const float*)d_in[2];
    const float* w_u    = (const float*)d_in[3];
    const float* b_u    = (const float*)d_in[4];
    const float* w_d    = (const float*)d_in[5];
    const float* b_d    = (const float*)d_in[6];
    const float* w_l    = (const float*)d_in[7];
    const float* b_l    = (const float*)d_in[8];
    const float* w_r    = (const float*)d_in[9];
    const float* b_r    = (const float*)d_in[10];
    const float* w_end  = (const float*)d_in[11];
    const float* b_end  = (const float*)d_in[12];

    float* Y   = (float*)d_ws;          // (16, 32, 256, 256) f32 intermediate
    float* Out = (float*)d_out;         // (16, 1, 256, 256)  f32

    const int pix_blocks = (Bn * HWSZ) / 256;   // 4096

    init_conv_kernel<<<pix_blocks, 256, 0, stream>>>(X, w_init, b_init, Y);

    // scan up   (axis=H, reverse): scan stride = W, conv along W (stride 1)
    scan_kernel<<<Bn, 512, 0, stream>>>(Y, w_u, b_u, 256, 1, 1);
    // scan down (axis=H, forward)
    scan_kernel<<<Bn, 512, 0, stream>>>(Y, w_d, b_d, 256, 1, 0);
    // scan left (axis=W, reverse): scan stride = 1, conv along H (stride W)
    scan_kernel<<<Bn, 512, 0, stream>>>(Y, w_l, b_l, 1, 256, 1);
    // scan right (axis=W, forward)
    scan_kernel<<<Bn, 512, 0, stream>>>(Y, w_r, b_r, 1, 256, 0);

    end_conv_kernel<<<pix_blocks, 256, 0, stream>>>(Y, w_end, b_end, Out);
}
